// DynTGCN_11982958756054
// MI455X (gfx1250) — compile-verified, matrix operand order checked by reading, not run on hardware
//
#include <hip/hip_runtime.h>
#include <hip/hip_bf16.h>
#include <math.h>

// ---------------- problem constants (match reference) ----------------
#define NN   50000   // nodes
#define FF   24      // in/out features
#define HH   128     // hidden
#define TT   12      // encoder steps
#define EE   800000  // edges
#define HOR  6       // decode horizon

typedef __bf16 bf16_t;
typedef bf16_t v16bf __attribute__((ext_vector_type(16)));
typedef float  v8f   __attribute__((ext_vector_type(8)));

// ---------------- small helper kernels ----------------
__global__ void zero_f32_kernel(float* p, size_t n) {
  size_t i = (size_t)blockIdx.x * blockDim.x + threadIdx.x;
  size_t st = (size_t)gridDim.x * blockDim.x;
  for (; i < n; i += st) p[i] = 0.0f;
}

__global__ void zero_u8_kernel(unsigned char* p, int n) {
  int i = blockIdx.x * blockDim.x + threadIdx.x;
  if (i < n) p[i] = 0;
}

__global__ void deg_init_kernel(float* deg, int n) {
  int i = blockIdx.x * blockDim.x + threadIdx.x;
  if (i < n) deg[i] = 1.0f;   // self-loop contributes 1
}

__global__ void deg_scatter_kernel(const int* __restrict__ dst,
                                   const float* __restrict__ ew,
                                   float* __restrict__ deg, int ne) {
  int i = blockIdx.x * blockDim.x + threadIdx.x;
  int st = gridDim.x * blockDim.x;
  for (; i < ne; i += st) atomicAdd(&deg[dst[i]], ew[i]);
}

// xw = x @ W   (N x F) @ (F x H), one block per node, one thread per H column
__global__ __launch_bounds__(HH) void input_proj_kernel(
    const float* __restrict__ x, const float* __restrict__ W,
    float* __restrict__ xw) {
  __shared__ float xs[32];
  int node = blockIdx.x;
  int col  = threadIdx.x;
  if (threadIdx.x < FF) xs[threadIdx.x] = x[(size_t)node * FF + threadIdx.x];
  __syncthreads();
  float acc = 0.0f;
#pragma unroll
  for (int k = 0; k < FF; ++k) acc += xs[k] * W[k * HH + col];
  xw[(size_t)node * HH + col] = acc;
}

// agg[dst] += xw[src] * (rsqrt(deg[src]) * ew * rsqrt(deg[dst])); one block per edge
__global__ __launch_bounds__(HH) void edge_scatter_kernel(
    const int* __restrict__ src, const int* __restrict__ dst,
    const float* __restrict__ ew, const float* __restrict__ deg,
    const float* __restrict__ xw, float* __restrict__ agg) {
  int e = blockIdx.x;
  int s = src[e], d = dst[e];                       // broadcast scalar loads
  float coef = __frsqrt_rn(deg[s]) * ew[e] * __frsqrt_rn(deg[d]);
  int c = threadIdx.x;
  atomicAdd(&agg[(size_t)d * HH + c], xw[(size_t)s * HH + c] * coef);
}

// hprev = (mask & seen) ? h_store : 0
__global__ void prep_hprev_kernel(const float* __restrict__ hstore,
                                  const unsigned char* __restrict__ mask,
                                  const unsigned char* __restrict__ seen,
                                  float* __restrict__ hprev) {
  size_t n = (size_t)NN * HH;
  size_t i = (size_t)blockIdx.x * blockDim.x + threadIdx.x;
  size_t st = (size_t)gridDim.x * blockDim.x;
  for (; i < n; i += st) {
    int node = (int)(i >> 7);   // /HH
    hprev[i] = (mask[node] && seen[node]) ? hstore[i] : 0.0f;
  }
}

__global__ void seen_update_kernel(unsigned char* seen,
                                   const unsigned char* __restrict__ mask, int n) {
  int i = blockIdx.x * blockDim.x + threadIdx.x;
  if (i < n) seen[i] = seen[i] | mask[i];
}

// rb *= h   (forms h*r for the candidate gate's concat operand)
__global__ void hr_kernel(float* __restrict__ rb, const float* __restrict__ h) {
  size_t n = (size_t)NN * HH;
  size_t i = (size_t)blockIdx.x * blockDim.x + threadIdx.x;
  size_t st = (size_t)gridDim.x * blockDim.x;
  for (; i < n; i += st) rb[i] *= h[i];
}

// Pre-pack a weight matrix B[K][Ncols] (f32, row stride ldb) into bf16 WMMA
// B-fragment layout:  out[((ct*ksteps + kk)*32 + lane)*16 + j]
//   = B[kk*32 + (lane/16)*16 + j][ct*16 + lane%16]   (zero-padded cols)
__global__ void pack_b_kernel(const float* __restrict__ B, int ldb,
                              int K, int Ncols, int NT,
                              bf16_t* __restrict__ out) {
  int ksteps = K >> 5;
  int total = NT * ksteps * 32 * 16;
  int i = blockIdx.x * blockDim.x + threadIdx.x;
  int st = gridDim.x * blockDim.x;
  for (; i < total; i += st) {
    int j    = i & 15;
    int lane = (i >> 4) & 31;
    int kk   = (i >> 9) % ksteps;
    int ct   = (i >> 9) / ksteps;
    int k    = kk * 32 + (lane >> 4) * 16 + j;
    int col  = ct * 16 + (lane & 15);
    out[i] = (bf16_t)((col < Ncols) ? B[(size_t)k * ldb + col] : 0.0f);
  }
}

// ---------------- generic fused WMMA GEMM ----------------
// out[M x Ncols] = epi( A[M x K] @ B[K x Ncols] + bias )
//   A column k:  k <  K1 : gcnMode ? (agg + xw*(1/deg) + biasA) : A1
//                k >= K1 : A2 (concat operand), if A2 != null
// B comes pre-packed in bf16 fragment layout (Bp).
// epi: 0 = store, 1 = sigmoid, 2 = GRU update  h' = z*h + (1-z)*tanh(u)
//      (epi 2 writes only where mask[row] != 0, or always if mask == null)
// Tiling: 128 threads = 4 wave32; block tile 64(M) x NT*16(N).
template <int NT>
__global__ __launch_bounds__(128) void wmma_gemm_kernel(
    const float* __restrict__ A1, const float* __restrict__ xw,
    const float* __restrict__ deg, const float* __restrict__ biasA,
    const float* __restrict__ A2, int K1,
    const bf16_t* __restrict__ Bp,
    const float* __restrict__ bias,
    int M, int K, int Ncols, int lda,
    int epi, const float* __restrict__ zb, const float* __restrict__ hprev,
    const unsigned char* __restrict__ mask,
    float* __restrict__ out, int ldo, int gcnMode) {
  __shared__ __align__(32) bf16_t As[64 * 32];   // [row][k]

  const int tid  = threadIdx.x;
  const int wave = tid >> 5;
  const int lane = tid & 31;
  const int rowBase = blockIdx.x * 64;

  // speculative prefetch of the first A tile (global_prefetch_b8)
  {
    int grow = rowBase + (tid >> 1);
    if (grow > M - 1) grow = M - 1;
    __builtin_prefetch(A1 + (size_t)grow * lda + (tid & 1) * 16, 0, 1);
  }

  v8f acc[NT];
  const v8f vzero = {0.f, 0.f, 0.f, 0.f, 0.f, 0.f, 0.f, 0.f};
#pragma unroll
  for (int ct = 0; ct < NT; ++ct) acc[ct] = vzero;

  const int ksteps = K >> 5;
  for (int kk = 0; kk < ksteps; ++kk) {
    const int k0 = kk * 32;
    __syncthreads();
    // stage A (64 x 32).  Region choice is uniform per k-step; rows are
    // clamped (not guarded): OOB rows feed WMMA but are never stored.
    if (A2 != nullptr && k0 >= K1) {
      const int kb = k0 - K1;
#pragma unroll
      for (int u = 0; u < 16; ++u) {
        int l = tid + u * 128;
        int row = l >> 5, k = l & 31;
        int grow = rowBase + row; if (grow > M - 1) grow = M - 1;
        As[row * 32 + k] = (bf16_t)A2[(size_t)grow * lda + kb + k];
      }
    } else if (gcnMode) {
#pragma unroll
      for (int u = 0; u < 16; ++u) {
        int l = tid + u * 128;
        int row = l >> 5, k = l & 31;
        int grow = rowBase + row; if (grow > M - 1) grow = M - 1;
        size_t ix = (size_t)grow * lda + k0 + k;
        As[row * 32 + k] =
            (bf16_t)(A1[ix] + xw[ix] * (1.0f / deg[grow]) + biasA[k0 + k]);
      }
    } else {
#pragma unroll
      for (int u = 0; u < 16; ++u) {
        int l = tid + u * 128;
        int row = l >> 5, k = l & 31;
        int grow = rowBase + row; if (grow > M - 1) grow = M - 1;
        As[row * 32 + k] = (bf16_t)A1[(size_t)grow * lda + k0 + k];
      }
    }
    __syncthreads();

    // A fragment: row = wave*16 + lane%16, K-halfblock by lane/16 (32B LDS read)
    const int kh = (lane >> 4) * 16;
    const int ar = wave * 16 + (lane & 15);
    v16bf afrag = *(const v16bf*)(As + ar * 32 + kh);
    // B fragments straight from packed global (32B per lane, L2-resident)
    const bf16_t* bbase = Bp + ((size_t)kk * 32 + lane) * 16;
#pragma unroll
    for (int ct = 0; ct < NT; ++ct) {
      v16bf bfrag = *(const v16bf*)(bbase + (size_t)ct * ksteps * 512);
      acc[ct] = __builtin_amdgcn_wmma_f32_16x16x32_bf16(
          false, afrag, false, bfrag, (short)0, acc[ct], false, false);
    }
  }

  // epilogue: C/D layout — vgpr v: rows v (lanes 0-15) / v+8 (lanes 16-31)
  const int colLane = lane & 15;
  const int rowTop  = rowBase + wave * 16 + (lane >> 4) * 8;
  if (epi == 1) {
#pragma unroll
    for (int ct = 0; ct < NT; ++ct) {
      int col = ct * 16 + colLane;
      if (col >= Ncols) continue;
      float bb = bias[col];
#pragma unroll
      for (int v = 0; v < 8; ++v) {
        int row = rowTop + v;
        if (row >= M) continue;
        float u = acc[ct][v] + bb;
        out[(size_t)row * ldo + col] = 1.0f / (1.0f + __expf(-u));
      }
    }
  } else if (epi == 2) {
#pragma unroll
    for (int ct = 0; ct < NT; ++ct) {
      int col = ct * 16 + colLane;
      if (col >= Ncols) continue;
      float bb = bias[col];
#pragma unroll
      for (int v = 0; v < 8; ++v) {
        int row = rowTop + v;
        if (row >= M) continue;
        float u = acc[ct][v] + bb;
        size_t ix = (size_t)row * HH + col;
        float z = zb[ix], hp = hprev[ix];
        float val = z * hp + (1.0f - z) * tanhf(u);
        if (mask == nullptr || mask[row]) out[(size_t)row * ldo + col] = val;
      }
    }
  } else {
#pragma unroll
    for (int ct = 0; ct < NT; ++ct) {
      int col = ct * 16 + colLane;
      if (col >= Ncols) continue;
      float bb = bias[col];
#pragma unroll
      for (int v = 0; v < 8; ++v) {
        int row = rowTop + v;
        if (row >= M) continue;
        out[(size_t)row * ldo + col] = acc[ct][v] + bb;
      }
    }
  }
}

// ---------------- host orchestration ----------------
extern "C" void kernel_launch(void* const* d_in, const int* in_sizes, int n_in,
                              void* d_out, int out_size, void* d_ws, size_t ws_size,
                              hipStream_t stream) {
  const float*         x_seq   = (const float*)d_in[0];
  const int*           ei_seq  = (const int*)d_in[1];
  const float*         ea_seq  = (const float*)d_in[2];
  const unsigned char* mask_sq = (const unsigned char*)d_in[3];  // jnp bool = u8
  const float* Wz  = (const float*)d_in[5];
  const float* Wr  = (const float*)d_in[6];
  const float* Wh  = (const float*)d_in[7];
  const float* bz  = (const float*)d_in[8];
  const float* br  = (const float*)d_in[9];
  const float* bh  = (const float*)d_in[10];
  const float* LWz = (const float*)d_in[11];
  const float* LWr = (const float*)d_in[12];
  const float* LWh = (const float*)d_in[13];
  const float* Lbz = (const float*)d_in[14];
  const float* Lbr = (const float*)d_in[15];
  const float* Lbh = (const float*)d_in[16];
  const float* headW = (const float*)d_in[17];
  const float* headb = (const float*)d_in[18];
  float* outp = (float*)d_out;

  // workspace carve
  const size_t NH = (size_t)NN * HH;
  float* ws      = (float*)d_ws;
  float* h_store = ws; ws += NH;
  float* hprev   = ws; ws += NH;
  float* xw      = ws; ws += NH;
  float* agg     = ws; ws += NH;
  float* zb      = ws; ws += NH;
  float* rb      = ws; ws += NH;
  float* deg     = ws; ws += NN;            // NN*4 bytes keeps 32B alignment
  bf16_t* LWpk[3];
  LWpk[0] = (bf16_t*)ws;                    // each 2H*H = 32768 bf16
  LWpk[1] = LWpk[0] + 32768;
  LWpk[2] = LWpk[1] + 32768;
  bf16_t* headWpk = LWpk[2] + 32768;        // 2*4*512 = 4096 bf16
  unsigned char* seen = (unsigned char*)(headWpk + 4096);

  // deterministic re-init + weight packing every call
  zero_f32_kernel<<<4096, 256, 0, stream>>>(h_store, NH);
  zero_u8_kernel<<<(NN + 255) / 256, 256, 0, stream>>>(seen, NN);
  const float* LWs[3] = {LWz, LWr, LWh};
  for (int g = 0; g < 3; ++g)
    pack_b_kernel<<<128, 256, 0, stream>>>(LWs[g], HH, 2 * HH, HH, 8, LWpk[g]);
  pack_b_kernel<<<16, 256, 0, stream>>>(headW, FF, HH, FF, 2, headWpk);

  const int Mb = (NN + 63) / 64;
  const float* Ws[3]  = {Wz, Wr, Wh};
  const float* bs[3]  = {bz, br, bh};
  const float* Lbs[3] = {Lbz, Lbr, Lbh};

  auto run_cell = [&](const float* x, const int* src, const int* dst,
                      const float* ew, const unsigned char* mask,
                      const float* hsrc, float* hdst, bool doDeg) {
    if (doDeg) {
      deg_init_kernel<<<(NN + 255) / 256, 256, 0, stream>>>(deg, NN);
      deg_scatter_kernel<<<2048, 256, 0, stream>>>(dst, ew, deg, EE);
    }
    for (int g = 0; g < 3; ++g) {
      input_proj_kernel<<<NN, HH, 0, stream>>>(x, Ws[g], xw);
      zero_f32_kernel<<<4096, 256, 0, stream>>>(agg, NH);
      edge_scatter_kernel<<<EE, HH, 0, stream>>>(src, dst, ew, deg, xw, agg);
      if (g == 2) hr_kernel<<<4096, 256, 0, stream>>>(rb, hsrc);  // rb = r * h
      const float* A2 = (g == 2) ? rb : hsrc;
      int    epi  = (g == 2) ? 2 : 1;
      float* oput = (g == 0) ? zb : (g == 1) ? rb : hdst;
      wmma_gemm_kernel<8><<<Mb, 128, 0, stream>>>(
          agg, xw, deg, bs[g], A2, HH, LWpk[g], Lbs[g],
          NN, 2 * HH, HH, HH, epi, zb, hsrc, mask, oput, HH, 1);
    }
  };

  // -------- encoder --------
  for (int t = 0; t < TT; ++t) {
    const unsigned char* mask = mask_sq + (size_t)t * NN;
    prep_hprev_kernel<<<4096, 256, 0, stream>>>(h_store, mask, seen, hprev);
    const int* src = ei_seq + (size_t)t * 2 * EE;
    const int* dst = src + EE;
    run_cell(x_seq + (size_t)t * NN * FF, src, dst, ea_seq + (size_t)t * EE,
             mask, hprev, h_store, true);
    seen_update_kernel<<<(NN + 255) / 256, 256, 0, stream>>>(seen, mask, NN);
  }

  // -------- decoder (reuses last graph; deg still holds step T-1) --------
  const unsigned char* mlast = mask_sq + (size_t)(TT - 1) * NN;
  prep_hprev_kernel<<<4096, 256, 0, stream>>>(h_store, mlast, seen, hprev);
  const int*   srcL = ei_seq + (size_t)(TT - 1) * 2 * EE;
  const int*   dstL = srcL + EE;
  const float* ewL  = ea_seq + (size_t)(TT - 1) * EE;
  float* hA = hprev;   // current hidden
  float* hB = h_store; // next hidden (ping-pong)
  for (int t = 0; t < HOR; ++t) {
    const float* xin = (t == 0) ? (x_seq + (size_t)(TT - 1) * NN * FF)
                                : (outp + (size_t)(t - 1) * NN * FF);
    run_cell(xin, srcL, dstL, ewL, nullptr, hA, hB, false);
    // y = h_new @ headW + headb  -> preds[t]  (also next step's input)
    wmma_gemm_kernel<2><<<Mb, 128, 0, stream>>>(
        hB, nullptr, nullptr, nullptr, nullptr, 0, headWpk, headb,
        NN, HH, FF, HH, 0, nullptr, nullptr, nullptr,
        outp + (size_t)t * NN * FF, FF, 0);
    float* tmp = hA; hA = hB; hB = tmp;
  }
}
